// Model_55164559950139
// MI455X (gfx1250) — matrix operand order repaired and checked by reading, not compiled, not run on hardware
//
#include <hip/hip_runtime.h>
#include <hip/hip_bf16.h>
#include <stdint.h>

typedef __bf16 bf16;
typedef __attribute__((ext_vector_type(16))) __bf16 v16bf;
typedef __attribute__((ext_vector_type(8)))  __bf16 v8bf;
typedef __attribute__((ext_vector_type(8)))  float  v8f;

#define B_   256
#define L_   200
#define IN_  512
#define H_   256
#define NG_  768                 // 3*H
#define ML_  ((size_t)B_ * L_)   // 51200 rows for the big GEMMs

// ---------------------------------------------------------------------------
// fp32 -> bf16 conversion (grid-stride)
// ---------------------------------------------------------------------------
__global__ __launch_bounds__(256) void k_cvt(const float* __restrict__ s,
                                             bf16* __restrict__ d, int n) {
  int stride = gridDim.x * blockDim.x;
  for (int i = blockIdx.x * blockDim.x + threadIdx.x; i < n; i += stride)
    d[i] = (bf16)s[i];
}

__device__ __forceinline__ v16bf cat8(v8bf lo, v8bf hi) {
  return __builtin_shufflevector(lo, hi, 0, 1, 2, 3, 4, 5, 6, 7,
                                 8, 9, 10, 11, 12, 13, 14, 15);
}

// A fragment (16x32 bf16, M x K): lane = hv*16 + lm holds row m=lm;
// elems 0..15 -> K=k0+hv*16+e (rowp already offset by hv*16)
__device__ __forceinline__ v16bf load_a_frag(const bf16* __restrict__ rowp, int k0) {
  v8bf a0 = *(const v8bf*)(rowp + k0);
  v8bf a1 = *(const v8bf*)(rowp + k0 + 8);
  return cat8(a0, a1);
}

// B fragment (32x16 bf16, K x N) from row-major W[N][K]:
// lane = hv*16 + lm holds col n=lm; elems 0..15 -> K = k0 + hv*16 + e
__device__ __forceinline__ v16bf load_b_frag(const bf16* __restrict__ W, int n,
                                             int ldk, int k0, int hv) {
  const bf16* p = W + (size_t)n * ldk + k0 + hv * 16;
  v8bf b0 = *(const v8bf*)(p);
  v8bf b1 = *(const v8bf*)(p + 8);
  return cat8(b0, b1);
}

// ---------------------------------------------------------------------------
// xi = A(bf16,[M,512]) @ W^T(bf16,[768,512]) + bias -> fp32 [M,768]
// B tile (64 rows x 512 K = 64KB) staged once per WG into LDS via
// global_load_async_to_lds_b128 with an additive-rotate chunk swizzle; A is
// register double-buffered. blockIdx.z selects fwd/rev weight set.
// ---------------------------------------------------------------------------
__global__ __launch_bounds__(256)
void k_gemm_xi(const bf16* __restrict__ A,
               const bf16* __restrict__ W0, const float* __restrict__ b0,
               float* __restrict__ C0,
               const bf16* __restrict__ W1, const float* __restrict__ b1,
               float* __restrict__ C1) {
  const int K = IN_;                         // 512 -> 1024B per row, 64 chunks
  __shared__ __align__(16) char bsm[64 * 1024];

  const bf16* W = blockIdx.z ? W1 : W0;
  const float* bias = blockIdx.z ? b1 : b0;
  float* C = blockIdx.z ? C1 : C0;

  int tid = threadIdx.x;
  int wv = tid >> 5;
  int lane = tid & 31;
  int lm = lane & 15;
  int hv = lane >> 4;

  int mrow = ((int)blockIdx.x * 8 + wv) * 16;
  int n0 = (int)blockIdx.y * 64;

  // ---- async-stage B tile: rows n0..n0+63, full K, swizzled chunks ----
  const char* Wbase = (const char*)(W + (size_t)n0 * K);
#pragma unroll
  for (int i = 0; i < 16; ++i) {
    int c = i * 256 + tid;                   // 0..4095 16B chunks
    int row = c >> 6;
    int ci = c & 63;
    unsigned ldso = (unsigned)(uintptr_t)(bsm + row * 1024 + (((ci + row) & 63) << 4));
    unsigned long long ga =
        (unsigned long long)(uintptr_t)(Wbase + row * 1024 + (ci << 4));
    asm volatile("global_load_async_to_lds_b128 %0, %1, off"
                 :: "v"(ldso), "v"(ga) : "memory");
  }
  asm volatile("s_wait_asynccnt 0x0" ::: "memory");

  v8f c[4];
#pragma unroll
  for (int j = 0; j < 4; ++j) {
    float bv = bias[n0 + j * 16 + lm];
#pragma unroll
    for (int r = 0; r < 8; ++r) c[j][r] = bv;
  }

  __syncthreads();  // B tile visible to all waves

  const bf16* Arow = A + (size_t)(mrow + lm) * K + hv * 16;
  v16bf a_cur = load_a_frag(Arow, 0);

  for (int k0 = 0; k0 < K; k0 += 32) {
    v16bf a_nxt = a_cur;
    if (k0 + 32 < K) {
      __builtin_prefetch((const void*)(Arow + k0 + 160), 0, 1);
      a_nxt = load_a_frag(Arow, k0 + 32);
    }
    int ci0 = (k0 >> 3) + 2 * hv;            // first of two 16B chunks
#pragma unroll
    for (int j = 0; j < 4; ++j) {
      int row = j * 16 + lm;
      const char* rb = bsm + row * 1024;
      v8bf b0v = *(const v8bf*)(rb + (((ci0 + row) & 63) << 4));
      v8bf b1v = *(const v8bf*)(rb + (((ci0 + 1 + row) & 63) << 4));
      v16bf b = cat8(b0v, b1v);
      c[j] = __builtin_amdgcn_wmma_f32_16x16x32_bf16(
          false, a_cur, false, b, (short)0, c[j], false, false);
    }
    a_cur = a_nxt;
  }

#pragma unroll
  for (int j = 0; j < 4; ++j) {
#pragma unroll
    for (int r = 0; r < 8; ++r) {
      size_t row = (size_t)mrow + 8 * hv + r;
      C[row * NG_ + n0 + j * 16 + lm] = c[j][r];
    }
  }
}

// ---------------------------------------------------------------------------
// GRU scan. grid = (B/16, 2 dirs), block = 256 (8 waves).
// Wave w owns cols w*32+s*16 of every gate block -> gate math fuses in regs.
// xi loads hoisted to step start (hidden behind the GEMM); w_hh B fragments
// register double-buffered across the 8 k-steps.
// ---------------------------------------------------------------------------
__global__ __launch_bounds__(256)
void k_gru_scan(const float* __restrict__ xi_f, const float* __restrict__ xi_r,
                const bf16* __restrict__ whh_f, const bf16* __restrict__ whh_r,
                const float* __restrict__ bhh_f, const float* __restrict__ bhh_r,
                bf16* __restrict__ out_bf,   // [B, L, 512], cols dir*256..+255
                float* __restrict__ out_f32, // same layout, optional
                int wf32) {
  const int HB = H_ + 8;  // padded bf16 row stride (16B pad: kills bank conflicts)
  __shared__ float hf[16 * H_];
  __shared__ bf16 hb[16 * (H_ + 8)];

  int dir = blockIdx.y;
  const float* xi = dir ? xi_r : xi_f;
  const bf16* Whh = dir ? whh_r : whh_f;
  const float* bhh = dir ? bhh_r : bhh_f;
  int brow0 = (int)blockIdx.x * 16;

  int tid = threadIdx.x;
  int wv = tid >> 5;
  int lane = tid & 31;
  int lm = lane & 15;
  int hv = lane >> 4;

  for (int i = tid; i < 16 * H_; i += 256) hf[i] = 0.f;
  for (int i = tid; i < 16 * HB; i += 256) hb[i] = (bf16)0.f;

  // bias preload: wave w, gate g, sub s -> column g*256 + w*32 + s*16 + lm
  float bv[3][2];
#pragma unroll
  for (int g = 0; g < 3; ++g)
#pragma unroll
    for (int s = 0; s < 2; ++s)
      bv[g][s] = bhh[g * H_ + wv * 32 + s * 16 + lm];

  const bf16* arow = hb + lm * HB + hv * 16;

  __syncthreads();

  for (int step = 0; step < L_; ++step) {
    int tt = dir ? (L_ - 1 - step) : step;

    // ---- hoisted xi loads: independent of h, hide behind the GEMM ----
    float xr_[2][8], xz_[2][8], xn_[2][8];
#pragma unroll
    for (int s = 0; s < 2; ++s) {
      int col = wv * 32 + s * 16 + lm;
#pragma unroll
      for (int r = 0; r < 8; ++r) {
        int m = r + 8 * hv;
        size_t xoff = ((size_t)(brow0 + m) * L_ + tt) * NG_ + col;
        xr_[s][r] = xi[xoff];
        xz_[s][r] = xi[xoff + H_];
        xn_[s][r] = xi[xoff + 2 * H_];
      }
    }

    v8f c[3][2];
#pragma unroll
    for (int g = 0; g < 3; ++g)
#pragma unroll
      for (int s = 0; s < 2; ++s)
#pragma unroll
        for (int r = 0; r < 8; ++r) c[g][s][r] = bv[g][s];

    // ---- gh = h @ Whh^T + b_hh, B double-buffered in registers ----
    v16bf bfr[2][6];
    auto loadB = [&](int k0, v16bf* dst) {
#pragma unroll
      for (int g = 0; g < 3; ++g)
#pragma unroll
        for (int s = 0; s < 2; ++s) {
          int n = g * H_ + wv * 32 + s * 16 + lm;
          dst[g * 2 + s] = load_b_frag(Whh, n, H_, k0, hv);
        }
    };
    loadB(0, bfr[0]);
#pragma unroll
    for (int kk = 0; kk < 8; ++kk) {
      int cur = kk & 1;
      if (kk < 7) loadB((kk + 1) * 32, bfr[cur ^ 1]);
      v16bf a = load_a_frag(arow, kk * 32);
#pragma unroll
      for (int g = 0; g < 3; ++g)
#pragma unroll
        for (int s = 0; s < 2; ++s)
          c[g][s] = __builtin_amdgcn_wmma_f32_16x16x32_bf16(
              false, a, false, bfr[cur][g * 2 + s], (short)0, c[g][s], false,
              false);
    }

    __syncthreads();  // all waves done reading hb before gate phase rewrites it

    // gate math in registers; lane owns rows m=r+8*hv, cols wv*32+s*16+lm
#pragma unroll
    for (int s = 0; s < 2; ++s) {
      int col = wv * 32 + s * 16 + lm;
#pragma unroll
      for (int r = 0; r < 8; ++r) {
        int m = r + 8 * hv;
        float hr = c[0][s][r];
        float hz = c[1][s][r];
        float hn = c[2][s][r];
        float hp = hf[m * H_ + col];

        float rg = 1.f / (1.f + __expf(-(xr_[s][r] + hr)));
        float zg = 1.f / (1.f + __expf(-(xz_[s][r] + hz)));
        float av = xn_[s][r] + rg * hn;
        float e2 = __expf(-2.f * av);
        float ng = (1.f - e2) / (1.f + e2);  // tanh
        float hnew = (1.f - zg) * ng + zg * hp;

        hf[m * H_ + col] = hnew;
        hb[m * HB + col] = (bf16)hnew;
        size_t oo = ((size_t)(brow0 + m) * L_ + tt) * (2 * H_) + dir * H_ + col;
        out_bf[oo] = (bf16)hnew;
        if (wf32) out_f32[oo] = hnew;
      }
    }
    __syncthreads();  // h published before next step's GEMM
  }
}

// ---------------------------------------------------------------------------
// Heads: y = out@fc1^T+b (raw), y2 = out@fc2^T+b (raw),
// y3 = broadcast(softmax(y)/10) + softmax(y2). One block per batch row.
// ---------------------------------------------------------------------------
__global__ __launch_bounds__(256)
void k_head(const float* __restrict__ outf,
            const float* __restrict__ fc1w, const float* __restrict__ fc1b,
            const float* __restrict__ fc2w, const float* __restrict__ fc2b,
            float* __restrict__ y, float* __restrict__ y2,
            float* __restrict__ y3) {
  __shared__ float ly[L_];
  __shared__ float ly2[L_ * 10];
  __shared__ float red[256];
  int b = blockIdx.x;
  int tid = threadIdx.x;

  for (int i = tid; i < L_ * 11; i += 256) {
    int t = i / 11, j = i % 11;
    const float* o = outf + ((size_t)b * L_ + t) * (2 * H_);
    const float* w = (j == 0) ? fc1w : fc2w + (size_t)(j - 1) * (2 * H_);
    float acc = (j == 0) ? fc1b[0] : fc2b[j - 1];
    for (int k = 0; k < 2 * H_; k += 4) {
      float4 ov = *(const float4*)(o + k);
      float4 wvv = *(const float4*)(w + k);
      acc += ov.x * wvv.x + ov.y * wvv.y + ov.z * wvv.z + ov.w * wvv.w;
    }
    if (j == 0) ly[t] = acc; else ly2[t * 10 + j - 1] = acc;
  }
  __syncthreads();

  float v = -1e30f;
  for (int i = tid; i < L_; i += 256) v = fmaxf(v, ly[i]);
  red[tid] = v; __syncthreads();
  for (int s = 128; s > 0; s >>= 1) {
    if (tid < s) red[tid] = fmaxf(red[tid], red[tid + s]);
    __syncthreads();
  }
  float m1 = red[0]; __syncthreads();
  v = 0.f;
  for (int i = tid; i < L_; i += 256) v += __expf(ly[i] - m1);
  red[tid] = v; __syncthreads();
  for (int s = 128; s > 0; s >>= 1) {
    if (tid < s) red[tid] += red[tid + s];
    __syncthreads();
  }
  float s1 = red[0]; __syncthreads();

  v = -1e30f;
  for (int i = tid; i < L_ * 10; i += 256) v = fmaxf(v, ly2[i]);
  red[tid] = v; __syncthreads();
  for (int s = 128; s > 0; s >>= 1) {
    if (tid < s) red[tid] = fmaxf(red[tid], red[tid + s]);
    __syncthreads();
  }
  float m2 = red[0]; __syncthreads();
  v = 0.f;
  for (int i = tid; i < L_ * 10; i += 256) v += __expf(ly2[i] - m2);
  red[tid] = v; __syncthreads();
  for (int s = 128; s > 0; s >>= 1) {
    if (tid < s) red[tid] += red[tid + s];
    __syncthreads();
  }
  float s2 = red[0]; __syncthreads();

  for (int i = tid; i < L_; i += 256) y[(size_t)b * L_ + i] = ly[i];
  for (int i = tid; i < L_ * 10; i += 256) {
    float v2 = ly2[i];
    y2[(size_t)b * L_ * 10 + i] = v2;
    float p = __expf(ly[i / 10] - m1) / s1 * 0.1f;
    y3[(size_t)b * L_ * 10 + i] = p + __expf(v2 - m2) / s2;
  }
}

// ---------------------------------------------------------------------------
extern "C" void kernel_launch(void* const* d_in, const int* in_sizes, int n_in,
                              void* d_out, int out_size, void* d_ws,
                              size_t ws_size, hipStream_t stream) {
  const float* x = (const float*)d_in[0];
  const float* w_ih[4] = {(const float*)d_in[1], (const float*)d_in[5],
                          (const float*)d_in[9], (const float*)d_in[13]};
  const float* w_hh[4] = {(const float*)d_in[2], (const float*)d_in[6],
                          (const float*)d_in[10], (const float*)d_in[14]};
  const float* b_ih[4] = {(const float*)d_in[3], (const float*)d_in[7],
                          (const float*)d_in[11], (const float*)d_in[15]};
  const float* b_hh[4] = {(const float*)d_in[4], (const float*)d_in[8],
                          (const float*)d_in[12], (const float*)d_in[16]};
  const float* fc1w = (const float*)d_in[17];
  const float* fc1b = (const float*)d_in[18];
  const float* fc2w = (const float*)d_in[19];
  const float* fc2b = (const float*)d_in[20];

  char* base = (char*)d_ws;
  size_t off = 0;
  auto take = [&](size_t bytes) -> void* {
    void* p = base + off;
    off += (bytes + 255) & ~(size_t)255;
    return p;
  };

  const int SZ_WIH = NG_ * IN_;        // 393216
  const int SZ_WHH = NG_ * H_;         // 196608
  const size_t SZ_X = ML_ * (2 * H_);  // 26.2M elems
  const size_t SZ_XI = ML_ * NG_;      // 39.3M elems

  bf16* wihb[4];
  bf16* whhb[4];
  for (int i = 0; i < 4; ++i) wihb[i] = (bf16*)take(SZ_WIH * sizeof(bf16));
  for (int i = 0; i < 4; ++i) whhb[i] = (bf16*)take(SZ_WHH * sizeof(bf16));
  bf16* xbf = (bf16*)take(SZ_X * sizeof(bf16));
  bf16* l0bf = (bf16*)take(SZ_X * sizeof(bf16));
  float* xif = (float*)take(SZ_XI * sizeof(float));
  float* xir = (float*)take(SZ_XI * sizeof(float));
  float* outf = (float*)take(SZ_X * sizeof(float));

  for (int i = 0; i < 4; ++i) {
    k_cvt<<<512, 256, 0, stream>>>(w_ih[i], wihb[i], SZ_WIH);
    k_cvt<<<512, 256, 0, stream>>>(w_hh[i], whhb[i], SZ_WHH);
  }
  k_cvt<<<8192, 256, 0, stream>>>(x, xbf, (int)SZ_X);

  dim3 ggrid(400, 12, 2);  // M/128 x N/64 x dirs

  // layer 0
  k_gemm_xi<<<ggrid, 256, 0, stream>>>(xbf, wihb[0], b_ih[0], xif,
                                       wihb[1], b_ih[1], xir);
  k_gru_scan<<<dim3(16, 2), 256, 0, stream>>>(xif, xir, whhb[0], whhb[1],
                                              b_hh[0], b_hh[1], l0bf, outf, 0);
  // layer 1
  k_gemm_xi<<<ggrid, 256, 0, stream>>>(l0bf, wihb[2], b_ih[2], xif,
                                       wihb[3], b_ih[3], xir);
  k_gru_scan<<<dim3(16, 2), 256, 0, stream>>>(xif, xir, whhb[2], whhb[3],
                                              b_hh[2], b_hh[3], xbf /*dummy*/,
                                              outf, 1);

  float* y = (float*)d_out;
  float* y2 = y + (size_t)B_ * L_;
  float* y3 = y2 + (size_t)B_ * L_ * 10;
  k_head<<<B_, 256, 0, stream>>>(outf, fc1w, fc1b, fc2w, fc2b, y, y2, y3);
}
